// DScriptLike_31009663877466
// MI455X (gfx1250) — matrix operand order repaired and checked by reading, not compiled, not run on hardware
//
#include <hip/hip_runtime.h>

typedef __attribute__((ext_vector_type(16))) _Float16 v16h;
typedef __attribute__((ext_vector_type(8)))  _Float16 v8h;
typedef __attribute__((ext_vector_type(8)))  float    v8f;

#define NN1 512
#define NN2 512
#define DIN 1280
#define DD  100
#define HH  20
#define KPAD 224
#define NPAD 32
#define NPIX (NN1*NN2)
#define EPSC 1e-5f
#define ASTRIDE 112

// workspace byte offsets (total ~12.0 MB)
#define OFF_A      0u
#define OFF_C      229376u     // 512*112*4
#define OFF_W2H    458752u     // 32*224*2 = 14336 B
#define OFF_SCALE1 473088u     // 20 f
#define OFF_SHIFT1 473168u     // 20 f
#define OFF_STATS  473248u     // sum1[20] ssq1[20] s2 ss2 scale2 shift2 = 44 f
#define OFF_CRAW   473600u     // 512*512*4
#define OFF_FC2H   1522176u    // 20*512*512*2

// Opaque |x|: clears the f32 sign bit via inline asm so the compiler can
// never form an fabs node (the v2f32 fabs custom lowering crashes gfx1250
// ISel, and InstCombine folds fmax(x,-x) back into fabs).
__device__ __forceinline__ float abs_opaque(float x) {
  float r;
  asm("v_and_b32 %0, 0x7fffffff, %1" : "=v"(r) : "v"(x));
  return r;
}

// ---------------- prep: zero stats, build padded f16 W2 ----------------
__global__ __launch_bounds__(256) void prep_kernel(const float* __restrict__ W2,
                                                   float* __restrict__ stats,
                                                   _Float16* __restrict__ w2h) {
  int t = blockIdx.x * 256 + threadIdx.x;
  if (t < 44) stats[t] = 0.f;
  if (t < NPAD * KPAD) {
    int n = t / KPAD, k = t % KPAD;
    float v = (n < HH && k < 2 * DD) ? W2[n * (2 * DD) + k] : 0.f;
    w2h[t] = (_Float16)v;
  }
}

// ---------------- projections a = x1 W1^T + b1, c = x2 W1^T + b1 ----------------
__global__ __launch_bounds__(128) void proj_kernel(const float* __restrict__ x1,
                                                   const float* __restrict__ x2,
                                                   const float* __restrict__ W1,
                                                   const float* __restrict__ b1,
                                                   float* __restrict__ aBuf,
                                                   float* __restrict__ cBuf) {
  __shared__ float sx[DIN];
  int row   = blockIdx.x & 511;
  int which = blockIdx.x >> 9;
  const float* x = which ? x2 : x1;
  float*       o = which ? cBuf : aBuf;
  for (int k = threadIdx.x; k < DIN; k += 128) sx[k] = x[row * DIN + k];
  __syncthreads();
  int h = threadIdx.x;
  if (h < ASTRIDE) {
    float acc = 0.f;
    if (h < DD) {
      const float* wrow = W1 + (size_t)h * DIN;
      for (int k = 0; k < DIN; ++k) acc = fmaf(sx[k], wrow[k], acc);
      acc += b1[h];
    }
    o[row * ASTRIDE + h] = acc;  // zero padding for h in [100,112)
  }
}

// ---------------- fused pairwise-features + fc2 via WMMA + BN1 stats ----------------
// block: 256 threads = 8 waves; wave w handles pairs (i, jb + 16w .. +15)
__global__ __launch_bounds__(256) void fc2_wmma_kernel(const float* __restrict__ aBuf,
                                                       const float* __restrict__ cBuf,
                                                       const _Float16* __restrict__ w2h,
                                                       const float* __restrict__ b2,
                                                       _Float16* __restrict__ fc2h,
                                                       float* __restrict__ stats) {
  __shared__ float sA[ASTRIDE];
  __shared__ float sC[128 * ASTRIDE];
  __shared__ float lsum[HH];
  __shared__ float lssq[HH];
  int tid = threadIdx.x;
  int i   = blockIdx.y;
  int jb  = blockIdx.x * 128;

  if (tid < ASTRIDE) sA[tid] = aBuf[i * ASTRIDE + tid];
  if (tid < HH) lsum[tid] = 0.f;
  if (tid >= HH && tid < 2 * HH) lssq[tid - HH] = 0.f;
  for (int t = tid; t < 128 * ASTRIDE; t += 256) sC[t] = cBuf[jb * ASTRIDE + t];
  __syncthreads();

  int wave = tid >> 5, lane = tid & 31;
  int m = lane & 15, sel = lane >> 4;
  const float* crow = &sC[(wave * 16 + m) * ASTRIDE];

  v8f acc0 = {};
  v8f acc1 = {};
#pragma unroll
  for (int kc = 0; kc < KPAD; kc += 32) {
    // A fragment (16x32 f16): lane sel=0 holds K kc+0..15, sel=1 holds kc+16..31
    union { v16h v; _Float16 h[16]; } Af;
#pragma unroll
    for (int t = 0; t < 16; ++t) {
      int k = kc + 16 * sel + t;
      float f;
      if (k < DD) {
        f = abs_opaque(sA[k] - crow[k]);
      } else if (k < 2 * DD) {
        f = sA[k - DD] * crow[k - DD];
      } else {
        f = 0.f;
      }
      Af.h[t] = (_Float16)f;
    }
    // B fragments (32x16 f16): lane group sel holds K kc+16*sel..+15 of column n = lane&15
    v16h B0 = *(const v16h*)(w2h + (size_t)m * KPAD + kc + 16 * sel);
    v16h B1 = *(const v16h*)(w2h + (size_t)(16 + m) * KPAD + kc + 16 * sel);
    acc0 = __builtin_amdgcn_wmma_f32_16x16x32_f16(false, Af.v, false, B0, (short)0, acc0, false, false);
    acc1 = __builtin_amdgcn_wmma_f32_16x16x32_f16(false, Af.v, false, B1, (short)0, acc1, false, false);
  }

  // D layout: element v of lane -> pair M = v + 8*sel, channel N = lane&15 (+16 for acc1)
  int jg = jb + wave * 16 + 8 * sel;  // global j for v = 0 (contiguous over v)
  {
    int h0 = m;
    float bb = b2[h0];
    float s = 0.f, q = 0.f;
    union { v8h v; _Float16 h[8]; } st;
#pragma unroll
    for (int v = 0; v < 8; ++v) {
      float d = acc0[v] + bb;
      s += d; q += d * d;
      st.h[v] = (_Float16)d;
    }
    *(v8h*)(fc2h + (size_t)h0 * NPIX + (size_t)i * NN2 + jg) = st.v;
    atomicAdd(&lsum[h0], s);
    atomicAdd(&lssq[h0], q);
  }
  {
    int h1 = 16 + m;
    if (h1 < HH) {
      float bb = b2[h1];
      float s = 0.f, q = 0.f;
      union { v8h v; _Float16 h[8]; } st;
#pragma unroll
      for (int v = 0; v < 8; ++v) {
        float d = acc1[v] + bb;
        s += d; q += d * d;
        st.h[v] = (_Float16)d;
      }
      *(v8h*)(fc2h + (size_t)h1 * NPIX + (size_t)i * NN2 + jg) = st.v;
      atomicAdd(&lsum[h1], s);
      atomicAdd(&lssq[h1], q);
    }
  }
  __syncthreads();
  if (tid < HH) {
    unsafeAtomicAdd(&stats[tid], lsum[tid]);
    unsafeAtomicAdd(&stats[HH + tid], lssq[tid]);
  }
}

// ---------------- BN1 finalize ----------------
__global__ void finalize1_kernel(const float* __restrict__ bn1_g, const float* __restrict__ bn1_b,
                                 float* __restrict__ scale1, float* __restrict__ shift1,
                                 const float* __restrict__ stats) {
  int h = threadIdx.x;
  if (h < HH) {
    float mu  = stats[h] / (float)NPIX;
    float var = stats[HH + h] / (float)NPIX - mu * mu;
    float sc  = bn1_g[h] * rsqrtf(var + EPSC);
    scale1[h] = sc;
    shift1[h] = bn1_b[h] - mu * sc;
  }
}

// ---------------- 7x7 conv (20->1), BN1+ReLU fused at load, BN2 stats ----------------
__global__ __launch_bounds__(256) void conv_kernel(const _Float16* __restrict__ fc2h,
                                                   const float* __restrict__ scale1,
                                                   const float* __restrict__ shift1,
                                                   const float* __restrict__ convw,
                                                   const float* __restrict__ convb,
                                                   float* __restrict__ craw,
                                                   float* __restrict__ stats) {
  __shared__ float tile[22 * 22];
  __shared__ float wsh[49];
  __shared__ float red[2];
  int tx = threadIdx.x & 15, ty = threadIdx.x >> 4;
  int x0 = blockIdx.x * 16, y0 = blockIdx.y * 16;
  float acc = 0.f;
  for (int h = 0; h < HH; ++h) {
    float s1 = scale1[h], sh1 = shift1[h];
    __syncthreads();
    if (threadIdx.x < 49) wsh[threadIdx.x] = convw[h * 49 + threadIdx.x];
    for (int t = threadIdx.x; t < 484; t += 256) {
      int r = t / 22, cc = t % 22;
      int gy = y0 - 3 + r, gx = x0 - 3 + cc;
      float v = 0.f;
      if (gy >= 0 && gy < NN1 && gx >= 0 && gx < NN2) {
        float f = (float)fc2h[(size_t)h * NPIX + (size_t)gy * NN2 + gx];
        v = fmaxf(f * s1 + sh1, 0.f);
      }
      tile[t] = v;
    }
    __syncthreads();
#pragma unroll
    for (int dy = 0; dy < 7; ++dy)
#pragma unroll
      for (int dx = 0; dx < 7; ++dx)
        acc = fmaf(tile[(ty + dy) * 22 + tx + dx], wsh[dy * 7 + dx], acc);
  }
  acc += convb[0];
  craw[(size_t)(y0 + ty) * NN2 + x0 + tx] = acc;
  if (threadIdx.x == 0) { red[0] = 0.f; red[1] = 0.f; }
  __syncthreads();
  atomicAdd(&red[0], acc);
  atomicAdd(&red[1], acc * acc);
  __syncthreads();
  if (threadIdx.x == 0) {
    unsafeAtomicAdd(&stats[40], red[0]);
    unsafeAtomicAdd(&stats[41], red[1]);
  }
}

// ---------------- BN2 finalize ----------------
__global__ void finalize2_kernel(const float* __restrict__ bn2_g, const float* __restrict__ bn2_b,
                                 float* __restrict__ stats) {
  if (threadIdx.x == 0) {
    float mu  = stats[40] / (float)NPIX;
    float var = stats[41] / (float)NPIX - mu * mu;
    float sc  = bn2_g[0] * rsqrtf(var + EPSC);
    stats[42] = sc;
    stats[43] = bn2_b[0] - mu * sc;
  }
}

// ---------------- apply BN2 + sigmoid -> output C plane ----------------
__global__ __launch_bounds__(256) void apply_kernel(const float* __restrict__ craw,
                                                    const float* __restrict__ stats,
                                                    float* __restrict__ outC) {
  int idx = blockIdx.x * 256 + threadIdx.x;
  if (idx < NPIX) {
    float v = craw[idx] * stats[42] + stats[43];
    outC[idx] = 1.f / (1.f + __expf(-v));
  }
}

// ---------------- maxpool 9/9 pad4 + column stats + final scalar ----------------
__global__ __launch_bounds__(256) void final_kernel(const float* __restrict__ Cs,
                                                    const float* __restrict__ gamma,
                                                    float* __restrict__ out0) {
  __shared__ float sB[57 * 57];
  __shared__ float sQ[57];
  __shared__ float sN[57];
  int tid = threadIdx.x;
  for (int cell = tid; cell < 57 * 57; cell += 256) {
    int r = cell / 57, c = cell % 57;
    int ylo = max(0, r * 9 - 4), yhi = min(511, r * 9 + 4);
    int xlo = max(0, c * 9 - 4), xhi = min(511, c * 9 + 4);
    float mx = -INFINITY;
    for (int y = ylo; y <= yhi; ++y)
      for (int x = xlo; x <= xhi; ++x)
        mx = fmaxf(mx, Cs[(size_t)y * NN2 + x]);
    sB[cell] = mx;
  }
  __syncthreads();
  if (tid < 57) {
    float s = 0.f;
    for (int r = 0; r < 57; ++r) s += sB[r * 57 + tid];
    float mean = s / 57.f;
    float v = 0.f;
    for (int r = 0; r < 57; ++r) { float d = sB[r * 57 + tid] - mean; v += d * d; }
    float sd = sqrtf(v / 56.f + EPSC);  // unbiased (ddof=1) + EPS
    float g = gamma[0];
    float qs = 0.f, qn = 0.f;
    for (int r = 0; r < 57; ++r) {
      float q = sB[r * 57 + tid] - mean - g * sd;
      if (q > 0.f) { qs += q; qn += 1.f; }
    }
    sQ[tid] = qs; sN[tid] = qn;
  }
  __syncthreads();
  if (tid == 0) {
    float qs = 0.f, qn = 0.f;
    for (int c = 0; c < 57; ++c) { qs += sQ[c]; qn += sN[c]; }
    float ph = qs / (qn + 1.f);
    ph = 1.f / (1.f + __expf(-ph));  // K_SIG=1, X0=0
    out0[0] = fminf(fmaxf(ph, 0.f), 1.f);
  }
}

extern "C" void kernel_launch(void* const* d_in, const int* in_sizes, int n_in,
                              void* d_out, int out_size, void* d_ws, size_t ws_size,
                              hipStream_t stream) {
  const float* x1    = (const float*)d_in[0];
  const float* x2    = (const float*)d_in[1];
  const float* W1    = (const float*)d_in[2];
  const float* b1    = (const float*)d_in[3];
  const float* W2    = (const float*)d_in[4];
  const float* b2    = (const float*)d_in[5];
  const float* bn1_g = (const float*)d_in[6];
  const float* bn1_b = (const float*)d_in[7];
  const float* convw = (const float*)d_in[8];
  const float* convb = (const float*)d_in[9];
  const float* bn2_g = (const float*)d_in[10];
  const float* bn2_b = (const float*)d_in[11];
  const float* gamma = (const float*)d_in[12];

  char* ws = (char*)d_ws;
  float*    aBuf   = (float*)(ws + OFF_A);
  float*    cBuf   = (float*)(ws + OFF_C);
  _Float16* w2h    = (_Float16*)(ws + OFF_W2H);
  float*    scale1 = (float*)(ws + OFF_SCALE1);
  float*    shift1 = (float*)(ws + OFF_SHIFT1);
  float*    stats  = (float*)(ws + OFF_STATS);
  float*    craw   = (float*)(ws + OFF_CRAW);
  _Float16* fc2h   = (_Float16*)(ws + OFF_FC2H);
  float*    out    = (float*)d_out;

  prep_kernel<<<28, 256, 0, stream>>>(W2, stats, w2h);
  proj_kernel<<<1024, 128, 0, stream>>>(x1, x2, W1, b1, aBuf, cBuf);
  dim3 g2(4, 512);
  fc2_wmma_kernel<<<g2, 256, 0, stream>>>(aBuf, cBuf, w2h, b2, fc2h, stats);
  finalize1_kernel<<<1, 32, 0, stream>>>(bn1_g, bn1_b, scale1, shift1, stats);
  dim3 g4(32, 32);
  conv_kernel<<<g4, 256, 0, stream>>>(fc2h, scale1, shift1, convw, convb, craw, stats);
  finalize2_kernel<<<1, 32, 0, stream>>>(bn2_g, bn2_b, stats);
  apply_kernel<<<(NPIX + 255) / 256, 256, 0, stream>>>(craw, stats, out + 1);
  final_kernel<<<1, 256, 0, stream>>>(out + 1, gamma, out);
}
